// SelfAttention_84920093377171
// MI455X (gfx1250) — matrix operand order repaired and checked by reading, not compiled, not run on hardware
//
#include <hip/hip_runtime.h>

// ---------------------------------------------------------------------------
// MI455X (gfx1250) self-attention: bf16 WMMA + async global->LDS staging.
// Shapes: B=64, N=512, E=1024, H=16, Dh=64.
// ---------------------------------------------------------------------------

typedef __attribute__((ext_vector_type(16))) __bf16 v16bf;
typedef __attribute__((ext_vector_type(8)))  float  v8f;

#define BB   64
#define NN   512
#define EE   1024
#define HH   16
#define DH   64

static __device__ __forceinline__ __bf16 f2bf(float f) {
    unsigned u = __builtin_bit_cast(unsigned, f);
    unsigned r = u + 0x7FFFu + ((u >> 16) & 1u);   // round-to-nearest-even
    return __builtin_bit_cast(__bf16, (unsigned short)(r >> 16));
}

// CDNA5 async global->LDS copy (ASYNCcnt-tracked).  The LDS destination VGPR
// holds the wave-relative LDS byte offset, which is exactly the low 32 bits of
// a generic pointer into __shared__ (flat->LDS maps addr[31:0], ISA 10.2).
static __device__ __forceinline__ void async_b128(void* lds_dst, const void* gsrc) {
    unsigned l = (unsigned)(unsigned long long)lds_dst;
    asm volatile("global_load_async_to_lds_b128 %0, %1, off"
                 :: "v"(l), "v"(gsrc) : "memory");
}
static __device__ __forceinline__ void wait_async0() {
    asm volatile("s_wait_asynccnt 0" ::: "memory");
}

// ---------------------------------------------------------------------------
// f32 -> bf16 elementwise conversion
// ---------------------------------------------------------------------------
__global__ void cvt_f32_bf16_kernel(const float* __restrict__ in,
                                    __bf16* __restrict__ out, int n) {
    int i = blockIdx.x * blockDim.x + threadIdx.x;
    if (i < n) out[i] = f2bf(in[i]);
}

// ---------------------------------------------------------------------------
// Tiled bf16 GEMM:  C[M,Nn] = A[M,K] @ Bm[K,Nn] + bias.
// Block tile 128x128, K-chunk 32, 8 waves; wave (wm,wn) owns 32x64 = 2x4
// WMMA 16x16 tiles.  A staged via async b128 copies (double-buffered);
// B staged transposed (sync, pipelined one buffer ahead) so both fragment
// gathers are K-contiguous b128 LDS loads.  Rows padded to 40 elems (80 B)
// -> 16 distinct banks, 16 B alignment preserved.
// ---------------------------------------------------------------------------
#define APAD 40
template <bool OUT_BF16>
__global__ __launch_bounds__(256)
void gemm_bf16_kernel(const __bf16* __restrict__ A,
                      const __bf16* __restrict__ Bm,
                      const float*  __restrict__ bias,
                      void* __restrict__ Cout,
                      int M, int K, int Nn) {
    __shared__ __bf16 sA[2][128][APAD];   // 20 KB
    __shared__ __bf16 sBt[2][128][APAD];  // 20 KB, transposed: [n][k]

    const int t    = threadIdx.x;
    const int lane = t & 31;
    const int wave = t >> 5;
    const int half = lane >> 4;
    const int l16  = lane & 15;
    const int m0   = blockIdx.y * 128;
    const int n0   = blockIdx.x * 128;
    const int wm   = wave & 3;       // 4 M sub-tiles of 32 rows
    const int wn   = wave >> 2;      // 2 N sub-tiles of 64 cols

    const int arow = t >> 1, akoff = (t & 1) * 16;   // A staging: 16 bf16/thread
    const int bn   = t & 127, bkq = (t >> 7) * 16;   // B staging: 16 bf16/thread

    v8f acc[2][4] = {};
    const int KT = K / 32;

    // prologue: stage buffer 0
    {
        const __bf16* src = A + (size_t)(m0 + arow) * K + akoff;
        async_b128(&sA[0][arow][akoff],     src);
        async_b128(&sA[0][arow][akoff + 8], src + 8);
        #pragma unroll
        for (int i = 0; i < 16; ++i)
            sBt[0][bn][bkq + i] = Bm[(size_t)(bkq + i) * Nn + (n0 + bn)];
    }

    for (int kt = 0; kt < KT; ++kt) {
        const int cur = kt & 1;
        wait_async0();
        __syncthreads();
        if (kt + 1 < KT) {           // stage next buffer; overlaps the WMMAs
            const int k0 = (kt + 1) * 32;
            const __bf16* src = A + (size_t)(m0 + arow) * K + (k0 + akoff);
            async_b128(&sA[cur ^ 1][arow][akoff],     src);
            async_b128(&sA[cur ^ 1][arow][akoff + 8], src + 8);
            #pragma unroll
            for (int i = 0; i < 16; ++i)
                sBt[cur ^ 1][bn][bkq + i] =
                    Bm[(size_t)(k0 + bkq + i) * Nn + (n0 + bn)];
        }

        v16bf afrag[2], bfrag[4];
        #pragma unroll
        for (int mi = 0; mi < 2; ++mi) {
            int row = wm * 32 + mi * 16 + l16;
            #pragma unroll
            for (int e = 0; e < 16; ++e) {
                int k = 16 * half + e;
                afrag[mi][e] = sA[cur][row][k];
            }
        }
        #pragma unroll
        for (int ni = 0; ni < 4; ++ni) {
            int n = wn * 64 + ni * 16 + l16;
            int kb = 16 * half;
            #pragma unroll
            for (int e = 0; e < 16; ++e) bfrag[ni][e] = sBt[cur][n][kb + e];
        }
        #pragma unroll
        for (int mi = 0; mi < 2; ++mi)
            #pragma unroll
            for (int ni = 0; ni < 4; ++ni)
                acc[mi][ni] = __builtin_amdgcn_wmma_f32_16x16x32_bf16(
                    false, afrag[mi], false, bfrag[ni],
                    (short)0, acc[mi][ni], false, false);
    }

    // Epilogue: C/D layout -> M = r + 8*half, N = lane&15
    #pragma unroll
    for (int mi = 0; mi < 2; ++mi)
        #pragma unroll
        for (int ni = 0; ni < 4; ++ni)
            #pragma unroll
            for (int r = 0; r < 8; ++r) {
                int m = m0 + wm * 32 + mi * 16 + r + 8 * half;
                int n = n0 + wn * 64 + ni * 16 + l16;
                float v = acc[mi][ni][r] + bias[n];
                if (OUT_BF16)
                    ((__bf16*)Cout)[(size_t)m * Nn + n] = f2bf(v);
                else
                    ((float*)Cout)[(size_t)m * Nn + n] = v;
            }
}

// ---------------------------------------------------------------------------
// Attention: one workgroup per (b, h, 64-row q block).  Dynamic LDS ~158 KB
// (CDNA5 320 KB/WGP makes the full 64x512 f32 score strip resident):
//   sS [64][516] f32 (pad->conflict-free P gathers), sQ [64][72] bf16,
//   sK [2][64][72] bf16 double-buffered async (reused as sVt[64][40] in PV),
//   softmax scratch.
// ---------------------------------------------------------------------------
#define SPAD     516
#define QPAD     72
#define VPAD     40
#define SOFF_Q   (64 * SPAD * 4)                 // 132096
#define SOFF_K   (SOFF_Q + 64 * QPAD * 2)        // 141312
#define KBUF     (64 * QPAD * 2)                 // 9216 per buffer
#define SOFF_MAX (SOFF_K + 2 * KBUF)             // 159744
#define SOFF_SUM (SOFF_MAX + 256)
#define SOFF_RED (SOFF_SUM + 256)
#define ATTN_LDS (SOFF_RED + 1024)               // 161280 bytes

__global__ __launch_bounds__(256)
void attn_kernel(const __bf16* __restrict__ qkv,
                 const int* __restrict__ adj,
                 __bf16* __restrict__ y) {
    extern __shared__ char smem[];
    float*  sS   = (float*)smem;                 // [64][SPAD]
    __bf16* sQ   = (__bf16*)(smem + SOFF_Q);     // [64][QPAD]
    __bf16* sK0  = (__bf16*)(smem + SOFF_K);     // [2][64][QPAD] / sVt[64][VPAD]
    float*  sMax = (float*)(smem + SOFF_MAX);
    float*  sSum = (float*)(smem + SOFF_SUM);
    float*  sRed = (float*)(smem + SOFF_RED);

    const int t    = threadIdx.x;
    const int lane = t & 31;
    const int wave = t >> 5;
    const int half = lane >> 4;
    const int l16  = lane & 15;

    const int qb = blockIdx.x & 7;          // q row block (8 per (b,h))
    const int bh = blockIdx.x >> 3;
    const int h  = bh & (HH - 1);
    const int b  = bh >> 4;
    const int n0 = qb * 64;

    const size_t rowbase = (size_t)b * NN;
    const int qcol = h * DH;
    const int kcol = EE + h * DH;
    const int vcol = 2 * EE + h * DH;

    // staging geometry: each thread moves one row-chunk of 16 bf16 (2x b128)
    const int srow = t >> 2;               // 0..63
    const int sd0  = (t & 3) * 16;         // 0,16,32,48

    // ---- prologue: async-load Q block and K block 0 ----
    {
        const __bf16* qsrc = qkv + (rowbase + n0 + srow) * (3 * EE) + qcol + sd0;
        async_b128(&sQ[srow * QPAD + sd0],     qsrc);
        async_b128(&sQ[srow * QPAD + sd0 + 8], qsrc + 8);
        const __bf16* ksrc = qkv + (rowbase + srow) * (3 * EE) + kcol + sd0;
        async_b128(&sK0[srow * QPAD + sd0],     ksrc);
        async_b128(&sK0[srow * QPAD + sd0 + 8], ksrc + 8);
    }

    // ---- pass 1: S = scale * Q K^T with adjacency mask ----
    const int qm  = wave & 3;               // q 16-row tile
    const int knp = wave >> 2;              // key tile pair
    for (int kb = 0; kb < 8; ++kb) {
        const int cur = kb & 1;
        wait_async0();
        __syncthreads();
        if (kb + 1 < 8) {                   // async-stage next K block
            __bf16* dst = sK0 + (cur ^ 1) * (64 * QPAD);
            const __bf16* ksrc =
                qkv + (rowbase + (kb + 1) * 64 + srow) * (3 * EE) + kcol + sd0;
            async_b128(&dst[srow * QPAD + sd0],     ksrc);
            async_b128(&dst[srow * QPAD + sd0 + 8], ksrc + 8);
        }
        const __bf16* sK = sK0 + cur * (64 * QPAD);

        #pragma unroll
        for (int j = 0; j < 2; ++j) {
            int kn = knp * 2 + j;
            v8f s = {};
            #pragma unroll
            for (int kc = 0; kc < 2; ++kc) {  // dh chunks of 32
                v16bf af, bf;
                int row = qm * 16 + l16;
                #pragma unroll
                for (int e = 0; e < 16; ++e) {
                    int k = kc * 32 + 16 * half + e;
                    af[e] = sQ[row * QPAD + k];
                }
                int n = kn * 16 + l16;        // B^T gather: sK[key][dh]
                #pragma unroll
                for (int e = 0; e < 16; ++e)
                    bf[e] = sK[n * QPAD + kc * 32 + 16 * half + e];
                s = __builtin_amdgcn_wmma_f32_16x16x32_bf16(
                        false, af, false, bf, (short)0, s, false, false);
            }
            #pragma unroll
            for (int r = 0; r < 8; ++r) {
                int m = qm * 16 + r + 8 * half;
                int col = kb * 64 + kn * 16 + l16;
                int qrow = n0 + m;
                bool keep = (adj[qrow * NN + col] != 0) || (qrow == col);
                sS[m * SPAD + col] = keep ? s[r] * 0.125f : -1.0e30f;
            }
        }
    }

    // ---- softmax over each of the 64 rows (4 threads per row) ----
    __syncthreads();
    {
        int row = t >> 2, q = t & 3;
        float* srowp = sS + row * SPAD + q * 128;
        float mx = -3.4e38f;
        for (int i = 0; i < 128; ++i) mx = fmaxf(mx, srowp[i]);
        sRed[t] = mx;
        __syncthreads();
        if (q == 0)
            sMax[row] = fmaxf(fmaxf(sRed[t], sRed[t + 1]),
                              fmaxf(sRed[t + 2], sRed[t + 3]));
        __syncthreads();
        float rm = sMax[row];
        float sum = 0.f;
        for (int i = 0; i < 128; ++i) {
            float e = __expf(srowp[i] - rm);
            srowp[i] = e;
            sum += e;
        }
        sRed[t] = sum;
        __syncthreads();
        if (q == 0)
            sSum[row] = 1.0f / (sRed[t] + sRed[t + 1] + sRed[t + 2] + sRed[t + 3]);
        __syncthreads();
        float inv = sSum[row];
        for (int i = 0; i < 128; ++i) srowp[i] *= inv;
    }

    // ---- pass 2: O = P V  (sVt staged transposed, sync) ----
    __bf16* sVt = sK0;                      // [64][VPAD] reuse
    const int dnp = wave >> 2;              // dh tile pair
    v8f oacc[2] = {};
    for (int vb = 0; vb < 16; ++vb) {       // key chunks of 32
        __syncthreads();
        #pragma unroll
        for (int i = 0; i < 8; ++i) {       // sVt[d][kk] = v[vb*32+kk][d]
            int idx = t * 8 + i;
            int d = idx >> 5, kk = idx & 31;
            sVt[d * VPAD + kk] =
                qkv[(rowbase + vb * 32 + kk) * (3 * EE) + vcol + d];
        }
        __syncthreads();

        v16bf af;
        {
            int row = qm * 16 + l16;
            #pragma unroll
            for (int e = 0; e < 16; ++e) {
                int k = 16 * half + e;
                af[e] = f2bf(sS[row * SPAD + vb * 32 + k]);
            }
        }
        #pragma unroll
        for (int j = 0; j < 2; ++j) {
            int dn = dnp * 2 + j;
            v16bf bf;
            int n = dn * 16 + l16;          // B^T gather: sVt[dh][key]
            #pragma unroll
            for (int e = 0; e < 16; ++e)
                bf[e] = sVt[n * VPAD + 16 * half + e];
            oacc[j] = __builtin_amdgcn_wmma_f32_16x16x32_bf16(
                          false, af, false, bf, (short)0, oacc[j], false, false);
        }
    }

    // ---- write y in [B, N, H, DH] order (bf16) ----
    #pragma unroll
    for (int j = 0; j < 2; ++j) {
        int dn = dnp * 2 + j;
        #pragma unroll
        for (int r = 0; r < 8; ++r) {
            int m = qm * 16 + r + 8 * half;
            int nrow = n0 + m;
            int d = dn * 16 + l16;
            y[(((size_t)b * NN + nrow) * HH + h) * DH + d] = f2bf(oacc[j][r]);
        }
    }
}

// ---------------------------------------------------------------------------
// Host launcher
// ---------------------------------------------------------------------------
extern "C" void kernel_launch(void* const* d_in, const int* in_sizes, int n_in,
                              void* d_out, int out_size, void* d_ws, size_t ws_size,
                              hipStream_t stream) {
    (void)in_sizes; (void)n_in; (void)out_size; (void)ws_size;
    const float* x     = (const float*)d_in[0];   // [B,N,E]
    const float* Wqkv  = (const float*)d_in[1];   // [E,3E]
    const float* bqkv  = (const float*)d_in[2];   // [3E]
    const float* Wproj = (const float*)d_in[3];   // [E,E]
    const float* bproj = (const float*)d_in[4];   // [E]
    const int*   adj   = (const int*)d_in[5];     // [N,N]

    const int M = BB * NN;                        // 32768

    // workspace carve-up (bf16 scratch, ~328 MB)
    char* ws = (char*)d_ws;
    __bf16* xb     = (__bf16*)(ws);                                   // 64 MB
    __bf16* wqkvb  = (__bf16*)(ws + (size_t)67108864);                //  6 MB
    __bf16* wprojb = (__bf16*)(ws + (size_t)73400320);                //  2 MB
    __bf16* qkv    = (__bf16*)(ws + (size_t)75497472);                // 192 MB
    __bf16* y      = (__bf16*)(ws + (size_t)276824064);               // 64 MB

    // 1) convert inputs to bf16
    {
        int n = M * EE;
        cvt_f32_bf16_kernel<<<(n + 255) / 256, 256, 0, stream>>>(x, xb, n);
    }
    {
        int n = EE * 3 * EE;
        cvt_f32_bf16_kernel<<<(n + 255) / 256, 256, 0, stream>>>(Wqkv, wqkvb, n);
    }
    {
        int n = EE * EE;
        cvt_f32_bf16_kernel<<<(n + 255) / 256, 256, 0, stream>>>(Wproj, wprojb, n);
    }

    // 2) QKV projection: [32768,1024] @ [1024,3072] -> bf16
    {
        dim3 grid(3 * EE / 128, M / 128);          // (24, 256)
        gemm_bf16_kernel<true><<<grid, 256, 0, stream>>>(
            xb, wqkvb, bqkv, (void*)qkv, M, EE, 3 * EE);
    }

    // 3) masked attention
    {
        hipFuncSetAttribute((const void*)attn_kernel,
                            hipFuncAttributeMaxDynamicSharedMemorySize, ATTN_LDS);
        attn_kernel<<<BB * HH * (NN / 64), 256, ATTN_LDS, stream>>>(qkv, adj, y);
    }

    // 4) output projection: [32768,1024] @ [1024,1024] -> f32 d_out
    {
        dim3 grid(EE / 128, M / 128);              // (8, 256)
        gemm_bf16_kernel<false><<<grid, 256, 0, stream>>>(
            y, wprojb, bproj, d_out, M, EE, EE);
    }
}
